// IndependentCNF_12412455486058
// MI455X (gfx1250) — matrix operand order repaired and checked by reading, not compiled, not run on hardware
//
#include <hip/hip_runtime.h>
#include <hip/hip_bf16.h>

typedef _Float16 v16h __attribute__((ext_vector_type(16)));
typedef _Float16 v8h  __attribute__((ext_vector_type(8)));
typedef _Float16 h2t  __attribute__((ext_vector_type(2)));
typedef float    v8f  __attribute__((ext_vector_type(8)));

#define WPB      4      // waves per block
#define HDIM     128
#define ASTRIDE  136    // padded f16 row stride in LDS (multiple of 8 -> 16B aligned rows)
#define AROWS    48     // 16 primal + 16 tangent-x0 + 16 tangent-x1
#define NSTEPS   16
#define SCF      12     // floats of per-element scratch
#define TIME_OFF 2.0f
#define LOG2PI   1.8378770664093453f

// ---------------- helpers ----------------

__device__ __forceinline__ void sp_sig(float u, float& sp, float& sg) {
  // stable softplus + sigmoid sharing one exp; v_rcp instead of IEEE divide
  float e = __expf(-fabsf(u));
  float r = __builtin_amdgcn_rcpf(1.0f + e);   // single v_rcp_f32 (TRANS, co-executes with WMMA)
  sg = (u >= 0.0f) ? r : (1.0f - r);
  sp = fmaxf(u, 0.0f) + __logf(1.0f + e);
}

// A-operand fragment (16x32 f16) from row-major LDS: two ds_load_b128 per lane.
__device__ __forceinline__ v16h load_afrag(const _Float16* act, int row, int kc, int hi) {
  const _Float16* p = act + row * ASTRIDE + kc * 32 + hi * 16;
  v8h lo = *(const v8h*)(p);        // K = kc*32 + hi*16 .. +7
  v8h hh = *(const v8h*)(p + 8);    // K = kc*32 + hi*16 + 8 .. +15
  v16h a;
  #pragma unroll
  for (int q = 0; q < 8; ++q) { a[q] = lo[q]; a[8 + q] = hh[q]; }
  return a;
}

// One hidden layer: primal + 2 tangents, 16x128 each, K=128 via WMMA.
// wg: f16 weights 128x128 row-major (row = output neuron, cols = K over h).
// tcol/bias: f32 t-column and bias in LDS.  act: per-wave LDS activations.
__device__ __forceinline__ void gemm_hidden(const _Float16* __restrict__ wg,
                                            const float* tcol, const float* bias,
                                            _Float16* act, const float* scw, int lane) {
  const int m  = lane & 15;
  const int hi = lane >> 4;
  v16h A0[4], A1[4], A2[4];
  #pragma unroll
  for (int kc = 0; kc < 4; ++kc) {     // load all A fragments before overwriting act
    A0[kc] = load_afrag(act, m,      kc, hi);
    A1[kc] = load_afrag(act, 16 + m, kc, hi);
    A2[kc] = load_afrag(act, 32 + m, kc, hi);
  }
  float tmv[8];
  #pragma unroll
  for (int i = 0; i < 8; ++i) tmv[i] = scw[(hi * 8 + i) * SCF + 2];
  #pragma unroll 2
  for (int nt = 0; nt < 8; ++nt) {
    const int nn = nt * 16 + m;        // this lane's output column
    const float w0 = tcol[nn];
    const float bb = bias[nn];
    v8f c0 = {}, c1 = {}, c2 = {};
    #pragma unroll
    for (int kc = 0; kc < 4; ++kc) {
      // B operand: 32 contiguous f16 of row nn (lane half selects K half)
      v16h b = *(const v16h*)(wg + nn * HDIM + kc * 32 + hi * 16);
      // reuse_b hint on the first two of the triplet: same B operand feeds the next instruction
      c0 = __builtin_amdgcn_wmma_f32_16x16x32_f16(false, A0[kc], false, b, (short)0, c0, false, true);
      c1 = __builtin_amdgcn_wmma_f32_16x16x32_f16(false, A1[kc], false, b, (short)0, c1, false, true);
      c2 = __builtin_amdgcn_wmma_f32_16x16x32_f16(false, A2[kc], false, b, (short)0, c2, false, false);
    }
    #pragma unroll
    for (int i = 0; i < 8; ++i) {      // C layout: vgpr i -> m = i + hi*8, n = lane&15
      const int mr = hi * 8 + i;
      float u = c0[i] + tmv[i] * w0 + bb;   // fold t-column + bias (K=129 concat)
      float sp, sg;
      sp_sig(u, sp, sg);
      act[mr * ASTRIDE + nn]        = (_Float16)sp;
      act[(16 + mr) * ASTRIDE + nn] = (_Float16)(sg * c1[i]);
      act[(32 + mr) * ASTRIDE + nn] = (_Float16)(sg * c2[i]);
    }
  }
}

// ---------------- prep: convert weights to f16 in d_ws ----------------

__global__ void cnf_prep(const float* __restrict__ W2, const float* __restrict__ W3,
                         const float* __restrict__ W4, _Float16* __restrict__ ws) {
  int i = blockIdx.x * 256 + threadIdx.x;
  if (i < HDIM * HDIM) {
    int n = i >> 7, k = i & 127;
    ws[i]               = (_Float16)W2[n * 129 + 1 + k];   // drop t-column (handled in f32)
    ws[HDIM * HDIM + i] = (_Float16)W3[n * 129 + 1 + k];
  }
  if (i < 16 * HDIM) {
    int n = i >> 7, k = i & 127;
    float v = (n < 2) ? W4[n * 129 + 1 + k] : 0.0f;        // zero-pad W4 to 16 rows
    ws[2 * HDIM * HDIM + i] = (_Float16)v;
  }
}

// ---------------- main kernel ----------------

__global__ __launch_bounds__(32 * WPB, 1)
void cnf_main(const float* __restrict__ ev, const float* __restrict__ xin,
              const float* __restrict__ maskp,
              const float* __restrict__ W1, const float* __restrict__ b1,
              const float* __restrict__ W2, const float* __restrict__ b2,
              const float* __restrict__ W3, const float* __restrict__ b3,
              const float* __restrict__ W4, const float* __restrict__ b4,
              const float* __restrict__ zm, const float* __restrict__ zls,
              const _Float16* __restrict__ wsh, float* __restrict__ outp) {
  __shared__ __attribute__((aligned(16))) _Float16 actAll[WPB][AROWS * ASTRIDE];
  __shared__ float sc[WPB][16 * SCF];
  __shared__ float sW1[HDIM * 3], sb1[HDIM], sw2t[HDIM], sb2[HDIM], sw3t[HDIM], sb3[HDIM];
  __shared__ float sW4t[2], sb4[2], szm[2], szls[2];

  const int tid  = threadIdx.x;
  const int lane = tid & 31;
  const int w    = tid >> 5;
  const int m    = lane & 15;
  const int hi   = lane >> 4;

  for (int i = tid; i < HDIM * 3; i += 32 * WPB) sW1[i] = W1[i];
  for (int i = tid; i < HDIM; i += 32 * WPB) {
    sb1[i]  = b1[i];
    sw2t[i] = W2[i * 129];
    sb2[i]  = b2[i];
    sw3t[i] = W3[i * 129];
    sb3[i]  = b3[i];
  }
  if (tid < 2) { sW4t[tid] = W4[tid * 129]; sb4[tid] = b4[tid]; szm[tid] = zm[tid]; szls[tid] = zls[tid]; }
  __syncthreads();

  _Float16* act = &actAll[w][0];
  float*    scw = &sc[w][0];
  const _Float16* w2g = wsh;
  const _Float16* w3g = wsh + HDIM * HDIM;
  const _Float16* w4g = wsh + 2 * HDIM * HDIM;

  const int e = blockIdx.x * (WPB * 16) + w * 16 + m;
  float t0 = 0.f, rat = 0.f, xc0 = 0.f, xc1 = 0.f, dlp = 0.f, msk = 0.f;
  if (hi == 0) {
    float tev = ev[e];
    t0  = tev + TIME_OFF;
    rat = -t0;                       // (t1 - t0), t1 = 0
    xc0 = xin[e * 2 + 0];
    xc1 = xin[e * 2 + 1];
    msk = maskp[e];
  }

  const float dsv = 1.0f / NSTEPS;

  #pragma unroll 1
  for (int step = 0; step < NSTEPS; ++step) {
    const float s = step * dsv;
    float pk0 = 0.f, pk1 = 0.f;           // previous-stage k (x)
    float ax0s = 0.f, ax1s = 0.f, als = 0.f;
    #pragma unroll 1
    for (int st = 0; st < 4; ++st) {
      if (hi == 0) {
        float coef = (st == 0) ? 0.f : ((st == 3) ? dsv : 0.5f * dsv);
        scw[m * SCF + 2] = t0 + (s + coef) * rat;   // t(s)
        scw[m * SCF + 3] = xc0 + coef * pk0;        // eval point
        scw[m * SCF + 4] = xc1 + coef * pk1;
      }
      // ---- layer 1 (3 -> 128) in VALU: primal + both tangent columns ----
      {
        float t   = scw[m * SCF + 2];
        float x0v = scw[m * SCF + 3];
        float x1v = scw[m * SCF + 4];
        _Float16* rowP = act + m * ASTRIDE;
        _Float16* rowA = act + (16 + m) * ASTRIDE;
        _Float16* rowB = act + (32 + m) * ASTRIDE;
        const int nb = hi * 64;
        #pragma unroll 4
        for (int j = 0; j < 64; j += 2) {
          int n0 = nb + j, n1 = n0 + 1;
          float u0 = sW1[n0 * 3] * t + sW1[n0 * 3 + 1] * x0v + sW1[n0 * 3 + 2] * x1v + sb1[n0];
          float u1 = sW1[n1 * 3] * t + sW1[n1 * 3 + 1] * x0v + sW1[n1 * 3 + 2] * x1v + sb1[n1];
          float sp0, sg0, sp1, sg1;
          sp_sig(u0, sp0, sg0);
          sp_sig(u1, sp1, sg1);
          h2t hp = { (_Float16)sp0, (_Float16)sp1 };
          h2t ha = { (_Float16)(sg0 * sW1[n0 * 3 + 1]), (_Float16)(sg1 * sW1[n1 * 3 + 1]) };
          h2t hb = { (_Float16)(sg0 * sW1[n0 * 3 + 2]), (_Float16)(sg1 * sW1[n1 * 3 + 2]) };
          *(h2t*)(rowP + n0) = hp;
          *(h2t*)(rowA + n0) = ha;
          *(h2t*)(rowB + n0) = hb;
        }
      }
      // ---- layers 2,3 (129 -> 128) via WMMA ----
      gemm_hidden(w2g, sw2t, sb2, act, scw, lane);
      gemm_hidden(w3g, sw3t, sb3, act, scw, lane);
      // ---- layer 4 (129 -> 2, padded to 16) via WMMA ----
      {
        v8f c0 = {}, c1 = {}, c2 = {};
        #pragma unroll
        for (int kc = 0; kc < 4; ++kc) {
          v16h a0 = load_afrag(act, m,      kc, hi);
          v16h a1 = load_afrag(act, 16 + m, kc, hi);
          v16h a2 = load_afrag(act, 32 + m, kc, hi);
          v16h b  = *(const v16h*)(w4g + m * HDIM + kc * 32 + hi * 16);
          c0 = __builtin_amdgcn_wmma_f32_16x16x32_f16(false, a0, false, b, (short)0, c0, false, true);
          c1 = __builtin_amdgcn_wmma_f32_16x16x32_f16(false, a1, false, b, (short)0, c1, false, true);
          c2 = __builtin_amdgcn_wmma_f32_16x16x32_f16(false, a2, false, b, (short)0, c2, false, false);
        }
        float tmv[8];
        #pragma unroll
        for (int i = 0; i < 8; ++i) tmv[i] = scw[(hi * 8 + i) * SCF + 2];
        if (m == 0) {        // column 0: dx0 (primal) and div from tangent-x0
          #pragma unroll
          for (int i = 0; i < 8; ++i) {
            int mr = hi * 8 + i;
            scw[mr * SCF + 5] = c0[i] + tmv[i] * sW4t[0] + sb4[0];
            scw[mr * SCF + 7] = c1[i];
          }
        } else if (m == 1) { // column 1: dx1 (primal) and div from tangent-x1
          #pragma unroll
          for (int i = 0; i < 8; ++i) {
            int mr = hi * 8 + i;
            scw[mr * SCF + 6] = c0[i] + tmv[i] * sW4t[1] + sb4[1];
            scw[mr * SCF + 8] = c2[i];
          }
        }
      }
      // ---- stage readback + RK4 accumulation ----
      if (hi == 0) {
        pk0 = scw[m * SCF + 5] * rat;
        pk1 = scw[m * SCF + 6] * rat;
        float pl = -(scw[m * SCF + 7] + scw[m * SCF + 8]) * rat;
        float wg = (st == 0 || st == 3) ? 1.f : 2.f;
        ax0s += wg * pk0;
        ax1s += wg * pk1;
        als  += wg * pl;
      }
    }
    if (hi == 0) {
      float c6 = dsv / 6.0f;
      xc0 += c6 * ax0s;
      xc1 += c6 * ax1s;
      dlp += c6 * als;
    }
  }

  if (hi == 0) {
    float iv0 = __expf(-szls[0]), iv1 = __expf(-szls[1]);
    float a0 = (xc0 - szm[0]) * iv0;
    float a1 = (xc1 - szm[1]) * iv1;
    float logpz = -0.5f * (a0 * a0 + 2.0f * szls[0] + LOG2PI)
                - 0.5f * (a1 * a1 + 2.0f * szls[1] + LOG2PI);
    outp[e] = (logpz - dlp) * msk;
  }
}

// ---------------- launch ----------------

extern "C" void kernel_launch(void* const* d_in, const int* in_sizes, int n_in,
                              void* d_out, int out_size, void* d_ws, size_t ws_size,
                              hipStream_t stream) {
  const float* ev   = (const float*)d_in[0];
  const float* xin  = (const float*)d_in[1];
  const float* mask = (const float*)d_in[2];
  const float* W1   = (const float*)d_in[3];
  const float* b1   = (const float*)d_in[4];
  const float* W2   = (const float*)d_in[5];
  const float* b2   = (const float*)d_in[6];
  const float* W3   = (const float*)d_in[7];
  const float* b3   = (const float*)d_in[8];
  const float* W4   = (const float*)d_in[9];
  const float* b4   = (const float*)d_in[10];
  const float* zm   = (const float*)d_in[11];
  const float* zls  = (const float*)d_in[12];
  _Float16* wsh = (_Float16*)d_ws;
  float* outp = (float*)d_out;

  const int B = in_sizes[0];                 // N*T = 131072
  cnf_prep<<<(HDIM * HDIM + 255) / 256, 256, 0, stream>>>(W2, W3, W4, wsh);
  const int blocks = B / (WPB * 16);
  cnf_main<<<blocks, 32 * WPB, 0, stream>>>(ev, xin, mask, W1, b1, W2, b2, W3, b3,
                                            W4, b4, zm, zls, wsh, outp);
}
